// MambaMIL_75746043232679
// MI455X (gfx1250) — matrix operand order repaired and checked by reading, not compiled, not run on hardware
//
#include <hip/hip_runtime.h>
#include <hip/hip_bf16.h>
#include <math.h>
#include <stdint.h>

#define L_SEQ  4096
#define DMODEL 512
#define DINNER 1024
#define DSTATE 16
#define DTRANK 32
#define INDIM  1024
#define NLAYERS 2
#define NCHUNK 64
#define CLEN   64

typedef __attribute__((ext_vector_type(16))) __bf16 v16bf;
typedef __attribute__((ext_vector_type(8)))  __bf16 v8bf;
typedef __attribute__((ext_vector_type(8)))  float  v8f;
typedef __attribute__((ext_vector_type(4)))  unsigned int u32x4;
typedef __attribute__((ext_vector_type(8)))  int  i32x8;
typedef __attribute__((ext_vector_type(4)))  int  i32x4;

// ---------------------------------------------------------------------------
// WMMA GEMM:  C[M,N] = epi( A[M,K](bf16, row stride lda) * W[N,K]^T(bf16) + bias ) + resid
// B tile (16 rows x K) staged into LDS once per block by the Tensor Data Mover.
// EPI: 0=none, 1=gelu(exact), 2=softplus, 3=tanh
// grid = (N/16, M/64), block = (32,4), dynamic LDS = 16*K*2 bytes.
// ---------------------------------------------------------------------------
template<int EPI>
__global__ void gemm_wmma(const __bf16* __restrict__ A, int lda,
                          const __bf16* __restrict__ W,
                          const float* __restrict__ bias,
                          const float* __restrict__ resid,
                          float* __restrict__ C,
                          __bf16* __restrict__ Cbf,
                          int M, int N, int K)
{
  extern __shared__ __bf16 lW[];               // [16][K] weight tile
  const int lane = threadIdx.x;                // 0..31
  const int l    = lane & 15;
  const int half = lane >> 4;                  // 0 or 1
  const int tn   = blockIdx.x * 16;
  const int tm   = blockIdx.y * 64 + threadIdx.y * 16;

  if (threadIdx.y == 0) {
    // --- TDM: DMA 16 x K bf16 tile (rows tn..tn+15 of W) into LDS ----------
    uint64_t g = (uint64_t)(uintptr_t)(W + (size_t)tn * K);   // tile start
    uint32_t ldsoff = (uint32_t)(uintptr_t)(&lW[0]);          // LDS byte offset
    u32x4 g0;
    g0[0] = 1u;                                               // count=1, user desc
    g0[1] = ldsoff;                                           // lds_addr
    g0[2] = (uint32_t)g;                                      // global_addr[31:0]
    g0[3] = (uint32_t)((g >> 32) & 0x01FFFFFFu) | (2u << 30); // addr[56:32] | type=2
    i32x8 g1;
    g1[0] = (int)(1u << 16);                                  // data_size=1 (2 bytes)
    g1[1] = (int)(((unsigned)K & 0xFFFFu) << 16);             // tensor_dim0[15:0]
    g1[2] = (int)((((unsigned)K >> 16) & 0xFFFFu) | (16u << 16)); // td0 hi | tensor_dim1=16
    g1[3] = (int)((unsigned)K << 16);                         // td1 hi=0 | tile_dim0=K
    g1[4] = 16;                                               // tile_dim1=16, tile_dim2=0
    g1[5] = K;                                                // tensor_dim0_stride[31:0]
    g1[6] = 0;                                                // stride0 hi | stride1 lo
    g1[7] = 0;
    i32x4 z4 = {0, 0, 0, 0};
#if __clang_major__ >= 23
    i32x8 z8 = {0, 0, 0, 0, 0, 0, 0, 0};
    __builtin_amdgcn_tensor_load_to_lds(g0, g1, z4, z4, z8, 0);
#else
    __builtin_amdgcn_tensor_load_to_lds(g0, g1, z4, z4, 0);
#endif
    __builtin_amdgcn_s_wait_tensorcnt(0);
  }
  __syncthreads();

  // per-lane fragment bases (ISA 16-bit WMMA layouts)
  const __bf16* arow = A + (size_t)(tm + l) * lda + half * 8;   // A: row tm+l
  const __bf16* wl   = lW + (size_t)l * K + half * 8;           // B: col tn+l (LDS)

  v8f acc = {};
  for (int k0 = 0; k0 < K; k0 += 32) {
    __builtin_prefetch(arow + k0 + 128, 0, 1);   // global_prefetch_b8
    v8bf a0 = *(const v8bf*)(arow + k0);         // K = kbase8 + 0..7
    v8bf a1 = *(const v8bf*)(arow + k0 + 16);    // K = kbase8 + 16..23
    v8bf b0 = *(const v8bf*)(wl + k0);           // K = kbase8 + 0..7
    v8bf b1 = *(const v8bf*)(wl + k0 + 16);      // K = kbase8 + 16..23
    v16bf af = __builtin_shufflevector(a0, a1, 0,1,2,3,4,5,6,7,8,9,10,11,12,13,14,15);
    v16bf bf = __builtin_shufflevector(b0, b1, 0,1,2,3,4,5,6,7,8,9,10,11,12,13,14,15);
    acc = __builtin_amdgcn_wmma_f32_16x16x32_bf16(false, af, false, bf,
                                                  (short)0, acc, false, false);
  }
  // C/D layout: VGPR r -> M = half*8 + r, N = lane%16
#pragma unroll
  for (int r = 0; r < 8; ++r) {
    int m = tm + half * 8 + r;
    int n = tn + l;
    float v = acc[r];
    if (bias) v += bias[n];
    if (EPI == 1)      v = 0.5f * v * (1.f + erff(v * 0.70710678118654752f));
    else if (EPI == 2) v = (v > 20.f) ? v : log1pf(__expf(v));
    else if (EPI == 3) v = tanhf(v);
    size_t o = (size_t)m * N + n;
    if (resid) v += resid[o];
    C[o] = v;
    if (Cbf) Cbf[o] = (__bf16)v;
  }
}

// ---------------------------------------------------------------------------
__global__ void cvt_bf16_k(const float* __restrict__ x, __bf16* __restrict__ y, int n)
{
  int i = blockIdx.x * 256 + threadIdx.x;
  if (i < n) y[i] = (__bf16)x[i];
}

// ---------------------------------------------------------------------------
__global__ void layernorm_k(const float* __restrict__ x,
                            const float* __restrict__ w,
                            const float* __restrict__ b,
                            float* __restrict__ y,
                            __bf16* __restrict__ ybf, int D)
{
  __shared__ float red[256];
  const int row = blockIdx.x, tid = threadIdx.x;
  const float* xr = x + (size_t)row * D;
  float s = 0.f;
  for (int i = tid; i < D; i += 256) s += xr[i];
  red[tid] = s; __syncthreads();
  for (int off = 128; off > 0; off >>= 1) { if (tid < off) red[tid] += red[tid + off]; __syncthreads(); }
  float mu = red[0] / D;
  __syncthreads();
  float vs = 0.f;
  for (int i = tid; i < D; i += 256) { float d0 = xr[i] - mu; vs += d0 * d0; }
  red[tid] = vs; __syncthreads();
  for (int off = 128; off > 0; off >>= 1) { if (tid < off) red[tid] += red[tid + off]; __syncthreads(); }
  float rstd = rsqrtf(red[0] / D + 1e-5f);
  for (int i = tid; i < D; i += 256) {
    float v = (xr[i] - mu) * rstd * w[i] + b[i];
    y[(size_t)row * D + i] = v;
    if (ybf) ybf[(size_t)row * D + i] = (__bf16)v;
  }
}

// ---------------------------------------------------------------------------
// depthwise causal conv (width 4) + SiLU; input = xm part of xz (stride 2*DINNER)
__global__ void conv_silu_k(const float* __restrict__ xz,
                            const float* __restrict__ cw,
                            const float* __restrict__ cb,
                            float* __restrict__ xc,
                            __bf16* __restrict__ xcb)
{
  int idx = blockIdx.x * 256 + threadIdx.x;      // t*DINNER + d
  int t = idx >> 10, d = idx & (DINNER - 1);
  float acc = cb[d];
#pragma unroll
  for (int j = 0; j < 4; ++j) {
    int tt = t - 3 + j;
    if (tt >= 0) acc = fmaf(cw[d * 4 + j], xz[(size_t)tt * (2 * DINNER) + d], acc);
  }
  float v = acc / (1.f + __expf(-acc));
  xc[idx] = v;
  xcb[idx] = (__bf16)v;
}

// ---------------------------------------------------------------------------
// Chunked parallel linear-recurrence scan. Pass 1: per-chunk transform (P,Q).
__global__ void scan_local_k(const float* __restrict__ dtv,
                             const float* __restrict__ xc,
                             const float* __restrict__ xdbl,   // [L,64]: dt|B|C
                             const float* __restrict__ A_log,
                             float* __restrict__ P, float* __restrict__ Q)
{
  __shared__ float lB[CLEN * DSTATE];
  const int c = blockIdx.x;
  const int d = blockIdx.y * 256 + threadIdx.x;
  for (int i = threadIdx.x; i < CLEN * DSTATE; i += 256) {
    int tt = i >> 4, s = i & 15;
    lB[i] = xdbl[(size_t)(c * CLEN + tt) * 64 + DTRANK + s];
  }
  __syncthreads();
  float a[16], b[16], Ad[16];
#pragma unroll
  for (int s = 0; s < 16; ++s) { Ad[s] = -__expf(A_log[d * 16 + s]); a[s] = 1.f; b[s] = 0.f; }
  for (int tt = 0; tt < CLEN; ++tt) {
    int t = c * CLEN + tt;
    float dt = dtv[(size_t)t * DINNER + d];
    float xv = xc [(size_t)t * DINNER + d];
#pragma unroll
    for (int s = 0; s < 16; ++s) {
      float dA = __expf(dt * Ad[s]);
      a[s] *= dA;
      b[s] = fmaf(dA, b[s], dt * lB[tt * 16 + s] * xv);
    }
  }
  size_t o = ((size_t)c * DINNER + d) * 16;
#pragma unroll
  for (int s = 0; s < 16; ++s) { P[o + s] = a[s]; Q[o + s] = b[s]; }
}

// Pass 2: prefix over chunks, parallel across all (d,s) pairs.
__global__ void scan_prefix_k(const float* __restrict__ P,
                              const float* __restrict__ Q,
                              float* __restrict__ hprev)
{
  int i = blockIdx.x * 256 + threadIdx.x;        // (d*16+s), 16384 total
  float h = 0.f;
  for (int c = 0; c < NCHUNK; ++c) {
    size_t o = (size_t)c * (DINNER * DSTATE) + i;
    hprev[o] = h;                                 // state entering chunk c
    h = fmaf(P[o], h, Q[o]);
  }
}

// Pass 3: replay within chunk from hprev, emit y = h.C + D*x, gate with silu(z)
__global__ void scan_apply_k(const float* __restrict__ dtv,
                             const float* __restrict__ xc,
                             const float* __restrict__ xdbl,
                             const float* __restrict__ A_log,
                             const float* __restrict__ hprev,
                             const float* __restrict__ Dv,
                             const float* __restrict__ xz,    // z at col offset DINNER
                             __bf16* __restrict__ yg)
{
  __shared__ float lB[CLEN * DSTATE];
  __shared__ float lC[CLEN * DSTATE];
  const int c = blockIdx.x;
  const int d = blockIdx.y * 256 + threadIdx.x;
  for (int i = threadIdx.x; i < CLEN * DSTATE; i += 256) {
    int tt = i >> 4, s = i & 15;
    lB[i] = xdbl[(size_t)(c * CLEN + tt) * 64 + DTRANK + s];
    lC[i] = xdbl[(size_t)(c * CLEN + tt) * 64 + DTRANK + DSTATE + s];
  }
  __syncthreads();
  float h[16], Ad[16];
  size_t ho = (size_t)c * (DINNER * DSTATE) + (size_t)d * 16;
#pragma unroll
  for (int s = 0; s < 16; ++s) { Ad[s] = -__expf(A_log[d * 16 + s]); h[s] = hprev[ho + s]; }
  float Dd = Dv[d];
  for (int tt = 0; tt < CLEN; ++tt) {
    int t = c * CLEN + tt;
    float dt = dtv[(size_t)t * DINNER + d];
    float xv = xc [(size_t)t * DINNER + d];
    float y = 0.f;
#pragma unroll
    for (int s = 0; s < 16; ++s) {
      float dA = __expf(dt * Ad[s]);
      h[s] = fmaf(dA, h[s], dt * lB[tt * 16 + s] * xv);
      y = fmaf(h[s], lC[tt * 16 + s], y);
    }
    y = fmaf(Dd, xv, y);
    float z = xz[(size_t)t * (2 * DINNER) + DINNER + d];
    yg[(size_t)t * DINNER + d] = (__bf16)(y * (z / (1.f + __expf(-z))));
  }
}

// ---------------------------------------------------------------------------
__global__ void attn_score_k(const float* __restrict__ a1,
                             const float* __restrict__ w2,
                             const float* __restrict__ b2,
                             float* __restrict__ score)
{
  int t = blockIdx.x * 256 + threadIdx.x;
  if (t >= L_SEQ) return;
  float s = b2[0];
  for (int i = 0; i < 128; ++i) s = fmaf(a1[(size_t)t * 128 + i], w2[i], s);
  score[t] = s;
}

__global__ void softmax_k(const float* __restrict__ score, float* __restrict__ w)
{
  __shared__ float red[1024];
  int tid = threadIdx.x;
  float m = -1e30f;
  for (int i = tid; i < L_SEQ; i += 1024) m = fmaxf(m, score[i]);
  red[tid] = m; __syncthreads();
  for (int off = 512; off > 0; off >>= 1) { if (tid < off) red[tid] = fmaxf(red[tid], red[tid + off]); __syncthreads(); }
  m = red[0]; __syncthreads();
  float s = 0.f;
  for (int i = tid; i < L_SEQ; i += 1024) s += __expf(score[i] - m);
  red[tid] = s; __syncthreads();
  for (int off = 512; off > 0; off >>= 1) { if (tid < off) red[tid] += red[tid + off]; __syncthreads(); }
  float inv = 1.f / red[0];
  for (int i = tid; i < L_SEQ; i += 1024) w[i] = __expf(score[i] - m) * inv;
}

__global__ void pool_k(const float* __restrict__ w, const float* __restrict__ hn,
                       float* __restrict__ pooled)
{
  int d = blockIdx.x * 256 + threadIdx.x;
  if (d >= DMODEL) return;
  float s = 0.f;
  for (int t = 0; t < L_SEQ; ++t) s = fmaf(w[t], hn[(size_t)t * DMODEL + d], s);
  pooled[d] = s;
}

__global__ void cls_k(const float* __restrict__ pooled, const float* __restrict__ cw,
                      const float* __restrict__ cb, float* __restrict__ out)
{
  int c = threadIdx.x;
  if (c < 2) {
    float s = cb[c];
    for (int d = 0; d < DMODEL; ++d) s = fmaf(pooled[d], cw[c * DMODEL + d], s);
    out[c] = s;
  }
}

// ---------------------------------------------------------------------------
extern "C" void kernel_launch(void* const* d_in, const int* in_sizes, int n_in,
                              void* d_out, int out_size, void* d_ws, size_t ws_size,
                              hipStream_t stream)
{
  (void)in_sizes; (void)n_in; (void)out_size; (void)ws_size;
  // pytree (alphabetical) flat order
  const float* attn_b1 = (const float*)d_in[0];
  const float* attn_b2 = (const float*)d_in[1];
  const float* attn_w1 = (const float*)d_in[2];
  const float* attn_w2 = (const float*)d_in[3];
  const float* cls_b   = (const float*)d_in[4];
  const float* cls_w   = (const float*)d_in[5];
  const float* fc1_b   = (const float*)d_in[6];
  const float* fc1_w   = (const float*)d_in[7];
  const float* norm_b  = (const float*)d_in[30];
  const float* norm_w  = (const float*)d_in[31];
  const float* x       = (const float*)d_in[32];

  float* ws = (float*)d_ws;
  size_t off = 0;
  auto allocf = [&](size_t n) { size_t a = (off + 3) & ~(size_t)3; off = a + n; return ws + a; };
  auto allocb = [&](size_t n) { size_t a = (off + 3) & ~(size_t)3; off = a + (n + 1) / 2; return (__bf16*)(ws + a); };

  float*  h      = allocf((size_t)L_SEQ * DMODEL);
  float*  xln    = allocf((size_t)L_SEQ * DMODEL);
  float*  xz     = allocf((size_t)L_SEQ * 2 * DINNER);
  float*  xc     = allocf((size_t)L_SEQ * DINNER);
  float*  xdbl   = allocf((size_t)L_SEQ * 64);
  float*  dtv    = allocf((size_t)L_SEQ * DINNER);
  float*  P      = allocf((size_t)NCHUNK * DINNER * DSTATE);
  float*  Q      = allocf((size_t)NCHUNK * DINNER * DSTATE);
  float*  hpre   = allocf((size_t)NCHUNK * DINNER * DSTATE);
  float*  hn     = allocf((size_t)L_SEQ * DMODEL);
  float*  a1     = allocf((size_t)L_SEQ * 128);
  float*  score  = allocf((size_t)L_SEQ);
  float*  smw    = allocf((size_t)L_SEQ);
  float*  pooled = allocf((size_t)DMODEL);
  __bf16* x_bf    = allocb((size_t)L_SEQ * INDIM);
  __bf16* xln_bf  = allocb((size_t)L_SEQ * DMODEL);
  __bf16* xc_bf   = allocb((size_t)L_SEQ * DINNER);
  __bf16* xdbl_bf = allocb((size_t)L_SEQ * 64);
  __bf16* yg_bf   = allocb((size_t)L_SEQ * DINNER);
  __bf16* hn_bf   = allocb((size_t)L_SEQ * DMODEL);
  __bf16* fc1w_bf = allocb((size_t)DMODEL * INDIM);
  __bf16* aw1_bf  = allocb((size_t)128 * DMODEL);
  __bf16* inw_bf  = allocb((size_t)2 * DINNER * DMODEL);
  __bf16* xpw_bf  = allocb((size_t)64 * DINNER);
  __bf16* dtw_bf  = allocb((size_t)DINNER * DTRANK);
  __bf16* outw_bf = allocb((size_t)DMODEL * DINNER);

  const dim3 gblk(32, 4, 1);
  auto cvt = [&](const float* src, __bf16* dst, size_t n) {
    cvt_bf16_k<<<(unsigned)((n + 255) / 256), 256, 0, stream>>>(src, dst, (int)n);
  };

  // one-time bf16 packing of stem/head weights + input
  cvt(x, x_bf, (size_t)L_SEQ * INDIM);
  cvt(fc1_w, fc1w_bf, (size_t)DMODEL * INDIM);
  cvt(attn_w1, aw1_bf, (size_t)128 * DMODEL);

  // stem: h = gelu(x @ fc1_w^T + fc1_b)
  gemm_wmma<1><<<dim3(DMODEL / 16, L_SEQ / 64), gblk, 16 * INDIM * 2, stream>>>(
      x_bf, INDIM, fc1w_bf, fc1_b, nullptr, h, nullptr, L_SEQ, DMODEL, INDIM);

  for (int layer = 0; layer < NLAYERS; ++layer) {
    int base = 8 + layer * 11;
    const float* A_log  = (const float*)d_in[base + 0];
    const float* Dv     = (const float*)d_in[base + 1];
    const float* conv_b = (const float*)d_in[base + 2];
    const float* conv_w = (const float*)d_in[base + 3];
    const float* dt_b   = (const float*)d_in[base + 4];
    const float* dt_w   = (const float*)d_in[base + 5];
    const float* in_w   = (const float*)d_in[base + 6];
    const float* ln_b   = (const float*)d_in[base + 7];
    const float* ln_w   = (const float*)d_in[base + 8];
    const float* out_w  = (const float*)d_in[base + 9];
    const float* xp_w   = (const float*)d_in[base + 10];

    cvt(in_w,  inw_bf,  (size_t)2 * DINNER * DMODEL);
    cvt(xp_w,  xpw_bf,  (size_t)64 * DINNER);
    cvt(dt_w,  dtw_bf,  (size_t)DINNER * DTRANK);
    cvt(out_w, outw_bf, (size_t)DMODEL * DINNER);

    layernorm_k<<<L_SEQ, 256, 0, stream>>>(h, ln_w, ln_b, xln, xln_bf, DMODEL);
    gemm_wmma<0><<<dim3((2 * DINNER) / 16, L_SEQ / 64), gblk, 16 * DMODEL * 2, stream>>>(
        xln_bf, DMODEL, inw_bf, nullptr, nullptr, xz, nullptr, L_SEQ, 2 * DINNER, DMODEL);
    conv_silu_k<<<(L_SEQ * DINNER) / 256, 256, 0, stream>>>(xz, conv_w, conv_b, xc, xc_bf);
    gemm_wmma<0><<<dim3(64 / 16, L_SEQ / 64), gblk, 16 * DINNER * 2, stream>>>(
        xc_bf, DINNER, xpw_bf, nullptr, nullptr, xdbl, xdbl_bf, L_SEQ, 64, DINNER);
    gemm_wmma<2><<<dim3(DINNER / 16, L_SEQ / 64), gblk, 16 * DTRANK * 2, stream>>>(
        xdbl_bf, 64, dtw_bf, dt_b, nullptr, dtv, nullptr, L_SEQ, DINNER, DTRANK);
    scan_local_k<<<dim3(NCHUNK, DINNER / 256), 256, 0, stream>>>(dtv, xc, xdbl, A_log, P, Q);
    scan_prefix_k<<<(DINNER * DSTATE) / 256, 256, 0, stream>>>(P, Q, hpre);
    scan_apply_k<<<dim3(NCHUNK, DINNER / 256), 256, 0, stream>>>(
        dtv, xc, xdbl, A_log, hpre, Dv, xz, yg_bf);
    gemm_wmma<0><<<dim3(DMODEL / 16, L_SEQ / 64), gblk, 16 * DINNER * 2, stream>>>(
        yg_bf, DINNER, outw_bf, nullptr, h, h, nullptr, L_SEQ, DMODEL, DINNER);
  }

  layernorm_k<<<L_SEQ, 256, 0, stream>>>(h, norm_w, norm_b, hn, hn_bf, DMODEL);
  gemm_wmma<3><<<dim3(128 / 16, L_SEQ / 64), gblk, 16 * DMODEL * 2, stream>>>(
      hn_bf, DMODEL, aw1_bf, attn_b1, nullptr, a1, nullptr, L_SEQ, 128, DMODEL);
  attn_score_k<<<L_SEQ / 256, 256, 0, stream>>>(a1, attn_w2, attn_b2, score);
  softmax_k<<<1, 1024, 0, stream>>>(score, smw);
  pool_k<<<DMODEL / 256, 256, 0, stream>>>(smw, hn, pooled);
  cls_k<<<1, 32, 0, stream>>>(pooled, cls_w, cls_b, (float*)d_out);
}
